// TaLKConv_19104014533228
// MI455X (gfx1250) — matrix operand order repaired and checked by reading, not compiled, not run on hardware
//
#include <hip/hip_runtime.h>
#include <hip/hip_bf16.h>

// TaLK-conv style adaptive-span kernel for MI455X (gfx1250, wave32).
// Pipeline: [offsets GEMM via v_wmma_f32_16x16x32_f16] + [3-pass segmented
// cumsum] -> [coalesced float4 interpolated gather].

typedef __attribute__((ext_vector_type(16))) _Float16 v16h;
typedef __attribute__((ext_vector_type(8)))  float    v8f;

#define Tn   2048
#define Bn   8
#define Cn   1024
#define Hn   8
#define Rn   128           // Cn / Hn
#define BHn  64            // Bn * Hn
#define COLS 8192          // BHn * Rn  (also Bn*Cn)
#define Mg   16384         // Tn * Bn   (GEMM rows)
#define Kg   1024          // Cn        (GEMM K)
#define TCH  128           // cumsum chunk length along T
#define NCH  16            // Tn / TCH

// ---------------------------------------------------------------------------
// Kernel 1: off = sigmoid(x @ W_off + b_off) via WMMA.
// One wave computes one 16x16 output tile (16 rows of (t,b), all 16 cols).
// 8 waves / block of 256 threads; grid = Mg/16/8 = 128 blocks.
// ---------------------------------------------------------------------------
__global__ __launch_bounds__(256)
void talk_offsets_wmma(const float* __restrict__ x,
                       const float* __restrict__ W,
                       const float* __restrict__ bias,
                       float* __restrict__ aL,
                       float* __restrict__ aR) {
    __shared__ _Float16 Wlds[Kg * 16];          // 32 KB: full W_off as f16
    for (int i = threadIdx.x; i < Kg * 16; i += blockDim.x)
        Wlds[i] = (_Float16)W[i];
    __syncthreads();

    const int wave = threadIdx.x >> 5;
    const int lane = threadIdx.x & 31;
    const int tile = blockIdx.x * 8 + wave;     // 0..1023
    const int half = lane >> 4;                 // 0: lanes 0-15, 1: lanes 16-31

    // A (16x32 f16): lane holds row (lane&15); K-base 0 or 8 by lane half.
    const int mrow  = tile * 16 + (lane & 15);
    const int kbA   = half * 8;
    // B (32x16 f16): lane holds col (lane&15); K-base 0 or 8 by lane half.
    const int ncol  = lane & 15;
    const int kbB   = half * 8;

    v8f c = {};
    const float* xrow = x + (size_t)mrow * Kg;
#pragma unroll 2
    for (int kk = 0; kk < Kg; kk += 32) {
        v16h a, b;
        // two contiguous 8-float segments of this lane's row (b128-friendly)
        const float* p0 = xrow + kk + kbA;       // K = kbA .. kbA+7
        const float* p1 = p0 + 16;               // K = kbA+16 .. kbA+23
#pragma unroll
        for (int j = 0; j < 8; ++j) a[j]     = (_Float16)p0[j];
#pragma unroll
        for (int j = 0; j < 8; ++j) a[8 + j] = (_Float16)p1[j];
#pragma unroll
        for (int j = 0; j < 16; ++j) b[j] = Wlds[(kk + kbB + (j < 8 ? j : j + 8)) * 16 + ncol];
        c = __builtin_amdgcn_wmma_f32_16x16x32_f16(
                /*neg_a=*/false, a, /*neg_b=*/false, b,
                /*c_mod=*/(short)0, c, /*reuse_a=*/false, /*reuse_b=*/false);
    }

    // D layout: VGPR v -> M = v + 8*half, N = lane&15.
    const float bv = bias[ncol];
    // branchless output select: n<8 -> aL, n>=8 -> aR (cndmask, no exec games)
    float* const dst = (ncol < Hn) ? aL : aR;
    const int n7 = ncol & 7;
#pragma unroll
    for (int v = 0; v < 8; ++v) {
        const int m    = tile * 16 + half * 8 + v;   // flat (t*Bn + b)
        const int t    = m >> 3;
        const int bidx = m & 7;
        // sigmoid via v_rcp_f32 (no IEEE divide sequence)
        const float sig = __builtin_amdgcn_rcpf(1.0f + __expf(-(c[v] + bv)));
        dst[t * BHn + bidx * Hn + n7] = sig;
    }
}

// ---------------------------------------------------------------------------
// Cumsum of x*(1/3) along T, segmented 3-pass for parallelism.
// ---------------------------------------------------------------------------
__global__ __launch_bounds__(256)
void talk_cum_pass1(const float* __restrict__ x, float* __restrict__ cs) {
    const int idx = blockIdx.x * blockDim.x + threadIdx.x;   // 131072
    const int ch  = idx >> 13;
    const int col = idx & (COLS - 1);
    const float sc = 1.0f / 3.0f;
    const float* p = x + (size_t)ch * TCH * COLS + col;
    float acc = 0.0f;
#pragma unroll 4
    for (int i = 0; i < TCH; ++i) acc += p[(size_t)i * COLS] * sc;
    cs[ch * COLS + col] = acc;
}

__global__ __launch_bounds__(256)
void talk_cum_pass2(float* __restrict__ cs) {
    const int col = blockIdx.x * blockDim.x + threadIdx.x;   // 8192
    float acc = 0.0f;
#pragma unroll
    for (int ch = 0; ch < NCH; ++ch) {
        const float v = cs[ch * COLS + col];
        cs[ch * COLS + col] = acc;                            // exclusive prefix
        acc += v;
    }
}

__global__ __launch_bounds__(256)
void talk_cum_pass3(const float* __restrict__ x, const float* __restrict__ cs,
                    float* __restrict__ S) {
    const int idx = blockIdx.x * blockDim.x + threadIdx.x;   // 131072
    const int ch  = idx >> 13;
    const int col = idx & (COLS - 1);
    const float sc = 1.0f / 3.0f;
    float acc = cs[ch * COLS + col];
    const size_t base = (size_t)ch * TCH * COLS + col;
#pragma unroll 4
    for (int i = 0; i < TCH; ++i) {
        const size_t o = base + (size_t)i * COLS;
        if (i + 8 < TCH) __builtin_prefetch(&x[o + 8 * COLS], 0, 0);
        acc += x[o] * sc;
        S[o] = acc;
    }
}

// ---------------------------------------------------------------------------
// Kernel 3: out = interp(S, fR) - interp(S, fL).
// One wave per (t,bh); each lane owns 4 consecutive channels (float4 b128).
// Each row gather is a fully coalesced 512B wave transaction. Fully
// branchless: the S(-1)==0 case is handled by a 0/1 multiplicative mask.
// ---------------------------------------------------------------------------
__device__ __forceinline__ float4 interp4(const float* __restrict__ S,
                                          int base, float f) {
    const float ff = floorf(f);
    const int   i0 = (int)ff;
    const float fr = f - ff;
    const int i0c = min(max(i0, 0), Tn - 1);
    const int i1c = min(max(i0 + 1, 0), Tn - 1);
    const float m0 = (i0 >= 0) ? 1.0f : 0.0f;    // cndmask, not a branch
    const float4 u0 = *(const float4*)(S + (size_t)i0c * COLS + base);
    const float4 v1 = *(const float4*)(S + (size_t)i1c * COLS + base);
    float4 r;
    const float v0x = m0 * u0.x, v0y = m0 * u0.y;
    const float v0z = m0 * u0.z, v0w = m0 * u0.w;
    r.x = v0x + fr * (v1.x - v0x);
    r.y = v0y + fr * (v1.y - v0y);
    r.z = v0z + fr * (v1.z - v0z);
    r.w = v0w + fr * (v1.w - v0w);
    return r;
}

__global__ __launch_bounds__(256)
void talk_interp(const float* __restrict__ S, const float* __restrict__ aL,
                 const float* __restrict__ aR, float* __restrict__ out) {
    const int pair = blockIdx.x * 8 + (threadIdx.x >> 5);    // 0..131071
    const int lane = threadIdx.x & 31;
    const int t  = pair >> 6;
    const int bh = pair & (BHn - 1);

    const float tf  = (float)t;
    const float aLv = aL[t * BHn + bh];
    const float aRv = aR[t * BHn + bh];
    const float lenL = 1.0f + aLv * fmaxf(tf - 1.0f, 0.0f);
    const float lenR = 1.0f + aRv * fmaxf(((float)(Tn - 1) - tf) - 1.0f, 0.0f);
    const float fL = fmaxf(tf - lenL - 1.0f, -1.0f);
    const float fR = fminf(tf + lenR, (float)(Tn - 1));

    const int base = bh * Rn + lane * 4;
    const float4 vR = interp4(S, base, fR);
    const float4 vL = interp4(S, base, fL);
    float4 o;
    o.x = vR.x - vL.x;  o.y = vR.y - vL.y;
    o.z = vR.z - vL.z;  o.w = vR.w - vL.w;
    // flat (t,bh,r) == flat (t,b,c) since bh*128+r == b*1024+c
    *(float4*)(out + (size_t)t * COLS + base) = o;
}

// ---------------------------------------------------------------------------
extern "C" void kernel_launch(void* const* d_in, const int* in_sizes, int n_in,
                              void* d_out, int out_size, void* d_ws, size_t ws_size,
                              hipStream_t stream) {
    const float* x    = (const float*)d_in[0];   // (T,B,C) f32
    const float* Woff = (const float*)d_in[1];   // (C,2H)  f32
    const float* boff = (const float*)d_in[2];   // (2H,)   f32
    float* out = (float*)d_out;                  // (T,B,C) f32

    // workspace layout
    float* S   = (float*)d_ws;                   // Tn*COLS       = 64 MB
    float* aL  = S  + (size_t)Tn * COLS;         // Tn*BHn floats = 512 KB
    float* aR  = aL + (size_t)Tn * BHn;          // 512 KB
    float* cs  = aR + (size_t)Tn * BHn;          // NCH*COLS      = 512 KB

    // 1) offsets GEMM (WMMA): 1024 tiles, 8 waves/block -> 128 blocks
    talk_offsets_wmma<<<Mg / 16 / 8, 256, 0, stream>>>(x, Woff, boff, aL, aR);

    // 2) segmented cumsum
    talk_cum_pass1<<<(NCH * COLS) / 256, 256, 0, stream>>>(x, cs);
    talk_cum_pass2<<<COLS / 256, 256, 0, stream>>>(cs);
    talk_cum_pass3<<<(NCH * COLS) / 256, 256, 0, stream>>>(x, cs, S);

    // 3) interpolated span difference: 131072 (t,bh) pairs, 8 waves/block
    talk_interp<<<(Tn * BHn) / 8, 256, 0, stream>>>(S, aL, aR, out);
}
